// MultiHeadAttention_19645180412316
// MI455X (gfx1250) — matrix operand order repaired and checked by reading, not compiled, not run on hardware
//
#include <hip/hip_runtime.h>
#include <math.h>

#define BATCH    4
#define SEQ      2048
#define D_MODEL  1024
#define N_HEAD   16
#define HEAD_DIM 64
#define AGENTS   256

typedef __attribute__((ext_vector_type(16))) __bf16 bf16x16;
typedef __attribute__((ext_vector_type(8)))  float  f32x8;
typedef __attribute__((ext_vector_type(4)))  unsigned int u32x4;
typedef __attribute__((ext_vector_type(8)))  int          i32x8;
typedef __attribute__((ext_vector_type(4)))  int          i32x4;

struct __align__(16) U4 { unsigned int x, y, z, w; };

union FragB {
  bf16x16        v;
  U4             q[2];
  unsigned short h[16];
};

__device__ __forceinline__ unsigned short f2bf(float f) {
  union { float f; unsigned u; } x; x.f = f;
  unsigned r = x.u + 0x7FFFu + ((x.u >> 16) & 1u);   // round-to-nearest-even
  return (unsigned short)(r >> 16);
}
__device__ __forceinline__ float bf2f(unsigned short h) {
  union { unsigned u; float f; } x; x.u = ((unsigned)h) << 16;
  return x.f;
}

__device__ __forceinline__ f32x8 wmma_bf16(const FragB& a, const FragB& b, f32x8 c) {
  return __builtin_amdgcn_wmma_f32_16x16x32_bf16(
      false, a.v, false, b.v, (short)0, c, false, false);
}

// ---------------------------------------------------------------------------
// Tensor Data Mover: 2D tile (tile_cols x tile_rows) of bf16 from global to
// LDS, with LDS padding of 16B after every 64B so that 32-element rows land
// on an 80-byte pitch (16B-aligned rows, conflict-free b128 reads).
// D# built per CDNA5 ISA ch.8 (group0: count/lds/global/type, group1: dims).
// ---------------------------------------------------------------------------
__device__ __forceinline__ void tdm_load_tile_bf16(
    unsigned lds_byte_off, const unsigned short* gsrc,
    unsigned tensor_rows, unsigned tensor_cols,
    unsigned tile_rows, unsigned tile_cols) {
  unsigned long long ga = (unsigned long long)gsrc;
  u32x4 g0;
  g0[0] = 1u;                                            // count=1 (valid D#)
  g0[1] = lds_byte_off;                                  // lds_addr
  g0[2] = (unsigned)(ga & 0xFFFFFFFFull);                // global_addr[31:0]
  g0[3] = (unsigned)((ga >> 32) & 0x1FFFFFFull) | (2u << 30);  // addr[56:32]|type=2
  i32x8 g1;
  g1[0] = (int)((1u << 16)          // data_size = 1 -> 2 bytes
              | (1u << 20)          // pad_enable
              | (3u << 22)          // pad_interval: 16 DWORDs (64B)
              | (3u << 25));        // pad_amount:   4 DWORDs (16B)
  g1[1] = (int)((tensor_cols & 0xFFFFu) << 16);                          // dim0 lo16
  g1[2] = (int)(((tensor_cols >> 16) & 0xFFFFu) | ((tensor_rows & 0xFFFFu) << 16));
  g1[3] = (int)(((tensor_rows >> 16) & 0xFFFFu) | ((tile_cols & 0xFFFFu) << 16));
  g1[4] = (int)(tile_rows & 0xFFFFu);                                    // tile_dim1, dim2=0
  g1[5] = (int)tensor_cols;                                              // dim0_stride lo32
  g1[6] = 0;
  g1[7] = 0;
  i32x4 z4 = {0, 0, 0, 0};
#if __has_include(<hip/amd_detail/amd_gfx1250_TDM.h>)
  i32x8 z8 = {0, 0, 0, 0, 0, 0, 0, 0};
  __builtin_amdgcn_tensor_load_to_lds(g0, g1, z4, z4, z8, 0);
#else
  __builtin_amdgcn_tensor_load_to_lds(g0, g1, z4, z4, 0);
#endif
}

// ---------------------------------------------------------------------------
// fp32 -> bf16 bulk convert (4 elems/thread)
// ---------------------------------------------------------------------------
__global__ void to_bf16(const float* __restrict__ in, unsigned short* __restrict__ out,
                        int n4) {
  int i = blockIdx.x * blockDim.x + threadIdx.x;
  if (i >= n4) return;
  const float4 f = ((const float4*)in)[i];
  unsigned lo = (unsigned)f2bf(f.x) | ((unsigned)f2bf(f.y) << 16);
  unsigned hi = (unsigned)f2bf(f.z) | ((unsigned)f2bf(f.w) << 16);
  ((uint2*)out)[i] = make_uint2(lo, hi);
}

// ---------------------------------------------------------------------------
// C[M,N] = X[M,K] @ W[N,K]^T + bias[N]; X,W bf16; C f32 or bf16.
// 128x128x32 tile, 8 waves: wave0 drives the TDM, everyone runs WMMA.
// ---------------------------------------------------------------------------
__global__ __launch_bounds__(256)
void gemm_bias_wmma(const unsigned short* __restrict__ Xb,
                    const unsigned short* __restrict__ Wb,
                    const float* __restrict__ bias, void* __restrict__ Cout,
                    int c_is_f32, int M, int N, int K) {
  __shared__ __align__(16) unsigned short As[128][40];  // 80B pitch (TDM-padded)
  __shared__ __align__(16) unsigned short Bs[128][40];

  const int tid   = threadIdx.x;
  const int lane  = tid & 31;
  const int wave  = tid >> 5;
  const int waveM = wave >> 2;    // 0..1
  const int waveN = wave & 3;     // 0..3
  const int m0    = blockIdx.y * 128;
  const int n0    = blockIdx.x * 128;
  const int half  = lane >> 4;
  const int l15   = lane & 15;

  const unsigned ldsA = (unsigned)(unsigned long long)(const void*)&As[0][0];
  const unsigned ldsB = (unsigned)(unsigned long long)(const void*)&Bs[0][0];

  f32x8 acc[4][2];
  for (int i = 0; i < 4; ++i)
    for (int j = 0; j < 2; ++j) acc[i][j] = {};

  for (int k0 = 0; k0 < K; k0 += 32) {
    if (wave == 0) {
      tdm_load_tile_bf16(ldsA, Xb + (size_t)m0 * K + k0, 128, (unsigned)K, 128, 32);
      tdm_load_tile_bf16(ldsB, Wb + (size_t)n0 * K + k0, 128, (unsigned)K, 128, 32);
      __builtin_amdgcn_s_wait_tensorcnt(0);
    } else if (wave == 1 && k0 + 32 < K) {   // global_prefetch_b8 hints for next tiles
      __builtin_prefetch(Xb + (size_t)(m0 + lane * 4) * K + k0 + 32, 0, 1);
      __builtin_prefetch(Wb + (size_t)(n0 + lane * 4) * K + k0 + 32, 0, 1);
    }
    __syncthreads();   // tiles visible to all waves

    FragB a[4], bf[2];
#pragma unroll
    for (int mt = 0; mt < 4; ++mt) {
      int row = waveM * 64 + mt * 16 + l15;
      int kb  = half * 8;                    // A: K {0..7,16..23}/{8..15,24..31}
      a[mt].q[0] = *(const U4*)&As[row][kb];
      a[mt].q[1] = *(const U4*)&As[row][kb + 16];
    }
#pragma unroll
    for (int nt = 0; nt < 2; ++nt) {
      int col = waveN * 32 + nt * 16 + l15;
      int kb  = half * 8;                    // B: K {0..7,16..23}/{8..15,24..31}
      bf[nt].q[0] = *(const U4*)&Bs[col][kb];
      bf[nt].q[1] = *(const U4*)&Bs[col][kb + 16];
    }
#pragma unroll
    for (int mt = 0; mt < 4; ++mt)
#pragma unroll
      for (int nt = 0; nt < 2; ++nt)
        acc[mt][nt] = wmma_bf16(a[mt], bf[nt], acc[mt][nt]);
    __syncthreads();   // done reading before wave0 overwrites
  }

#pragma unroll
  for (int mt = 0; mt < 4; ++mt)
#pragma unroll
    for (int nt = 0; nt < 2; ++nt) {
      int col  = n0 + waveN * 32 + nt * 16 + l15;
      float bv = bias[col];
      if (c_is_f32) {
        float* C = (float*)Cout;
#pragma unroll
        for (int j = 0; j < 8; ++j) {
          int row = m0 + waveM * 64 + mt * 16 + j + half * 8;
          C[(size_t)row * N + col] = acc[mt][nt][j] + bv;
        }
      } else {
        unsigned short* C = (unsigned short*)Cout;
#pragma unroll
        for (int j = 0; j < 8; ++j) {
          int row = m0 + waveM * 64 + mt * 16 + j + half * 8;
          C[(size_t)row * N + col] = f2bf(acc[mt][nt][j] + bv);
        }
      }
    }
}

// ---------------------------------------------------------------------------
// agent[b,h,a,k] = mean of 8 consecutive rows of projected Q (bf16 in/out)
// ---------------------------------------------------------------------------
__global__ void agent_pool(const unsigned short* __restrict__ Qb,
                           unsigned short* __restrict__ agent_b) {
  int idx = blockIdx.x * blockDim.x + threadIdx.x;   // B*H*A*64
  int kd  = idx & 63;
  int a   = (idx >> 6) & 255;
  int h   = (idx >> 14) & 15;
  int b   = idx >> 18;
  float s = 0.f;
#pragma unroll
  for (int t = 0; t < 8; ++t)
    s += bf2f(Qb[(size_t)(b * SEQ + a * 8 + t) * D_MODEL + h * HEAD_DIM + kd]);
  agent_b[idx] = f2bf(s * 0.125f);
}

// ---------------------------------------------------------------------------
// Stage 1: agent_v = softmax_S(agent @ kh^T * scale) @ vh   (all bf16 operands)
// One wave per (b,h,16-agent block); flash-style online softmax over S.
// ---------------------------------------------------------------------------
__global__ __launch_bounds__(32)
void stage1_agent_kv(const unsigned short* __restrict__ Kb,
                     const unsigned short* __restrict__ Vb,
                     const unsigned short* __restrict__ agent_b,
                     unsigned short* __restrict__ agent_vb) {
  __shared__ __align__(16) unsigned short Plds[16][40];
  const int lane = threadIdx.x;
  const int l15  = lane & 15, half = lane >> 4;
  const int a0   = blockIdx.x * 16;
  const int h    = blockIdx.y;
  const int b    = blockIdx.z;
  const float scale = 0.125f;

  const unsigned short* agBase = agent_b + ((size_t)(b * N_HEAD + h) * AGENTS) * HEAD_DIM;
  const unsigned short* KbH    = Kb + (size_t)b * SEQ * D_MODEL + h * HEAD_DIM;
  const unsigned short* VbH    = Vb + (size_t)b * SEQ * D_MODEL + h * HEAD_DIM;

  FragB aAg[2];
#pragma unroll
  for (int kf = 0; kf < 2; ++kf) {
    int row = a0 + l15;
    int kb  = kf * 32 + half * 8;
    const unsigned short* p = agBase + (size_t)row * HEAD_DIM + kb;
    aAg[kf].q[0] = *(const U4*)p;
    aAg[kf].q[1] = *(const U4*)(p + 16);
  }

  f32x8 O[4];
  for (int i = 0; i < 4; ++i) O[i] = {};
  float m[8], l[8];
#pragma unroll
  for (int j = 0; j < 8; ++j) { m[j] = -INFINITY; l[j] = 0.f; }

  for (int s0 = 0; s0 < SEQ; s0 += 32) {
    f32x8 sc0 = {}, sc1 = {};
#pragma unroll
    for (int kf = 0; kf < 2; ++kf) {
      FragB b0, b1;
      int kb = kf * 32 + half * 8;
      const unsigned short* p0 = KbH + (size_t)(s0 + l15) * D_MODEL + kb;
      const unsigned short* p1 = KbH + (size_t)(s0 + 16 + l15) * D_MODEL + kb;
      b0.q[0] = *(const U4*)p0; b0.q[1] = *(const U4*)(p0 + 16);
      b1.q[0] = *(const U4*)p1; b1.q[1] = *(const U4*)(p1 + 16);
      sc0 = wmma_bf16(aAg[kf], b0, sc0);
      sc1 = wmma_bf16(aAg[kf], b1, sc1);
    }

#pragma unroll
    for (int j = 0; j < 8; ++j) {
      float v0 = sc0[j] * scale, v1 = sc1[j] * scale;
      float cm = fmaxf(v0, v1);
      for (int msk = 1; msk < 16; msk <<= 1) cm = fmaxf(cm, __shfl_xor(cm, msk, 32));
      float mnew  = fmaxf(m[j], cm);
      float alpha = __expf(m[j] - mnew);
      float p0 = __expf(v0 - mnew), p1 = __expf(v1 - mnew);
      float rs = p0 + p1;
      for (int msk = 1; msk < 16; msk <<= 1) rs += __shfl_xor(rs, msk, 32);
      l[j] = l[j] * alpha + rs;
      m[j] = mnew;
#pragma unroll
      for (int nt = 0; nt < 4; ++nt) O[nt][j] *= alpha;
      int prow = j + half * 8;
      Plds[prow][l15]      = f2bf(p0);
      Plds[prow][16 + l15] = f2bf(p1);
    }
    __syncthreads();

    FragB pf;
    {
      int kb = half * 8;
      pf.q[0] = *(const U4*)&Plds[l15][kb];
      pf.q[1] = *(const U4*)&Plds[l15][kb + 16];
    }
#pragma unroll
    for (int nt = 0; nt < 4; ++nt) {
      FragB vb;
      int dimc  = nt * 16 + l15;
      int kbase = half * 8;
#pragma unroll
      for (int e = 0; e < 16; ++e)
        vb.h[e] = VbH[(size_t)(s0 + kbase + (e & 7) + (e >> 3) * 16) * D_MODEL + dimc];
      O[nt] = wmma_bf16(pf, vb, O[nt]);
    }
    __syncthreads();
  }

#pragma unroll
  for (int nt = 0; nt < 4; ++nt) {
    int dimc = nt * 16 + l15;
#pragma unroll
    for (int j = 0; j < 8; ++j) {
      int a = a0 + j + half * 8;
      agent_vb[((size_t)(b * N_HEAD + h) * AGENTS + a) * HEAD_DIM + dimc] =
          f2bf(O[nt][j] / l[j]);
    }
  }
}

// ---------------------------------------------------------------------------
// Stage 2: q_attn = softmax_A(qh @ agent^T * scale) -> d_out (f32)
//          out_h  = q_attn @ agent_v               -> bf16 workspace
// ---------------------------------------------------------------------------
__global__ __launch_bounds__(32)
void stage2_q_agent(const unsigned short* __restrict__ Qb,
                    const unsigned short* __restrict__ agent_b,
                    const unsigned short* __restrict__ agent_vb,
                    float* __restrict__ q_attn,
                    unsigned short* __restrict__ out_hb) {
  __shared__ __align__(16) unsigned short Plds[16][256];
  const int lane = threadIdx.x;
  const int l15  = lane & 15, half = lane >> 4;
  const int s0   = blockIdx.x * 16;
  const int h    = blockIdx.y;
  const int b    = blockIdx.z;
  const float scale = 0.125f;

  const unsigned short* agBase = agent_b  + ((size_t)(b * N_HEAD + h) * AGENTS) * HEAD_DIM;
  const unsigned short* avBase = agent_vb + ((size_t)(b * N_HEAD + h) * AGENTS) * HEAD_DIM;

  FragB qf[2];
#pragma unroll
  for (int kf = 0; kf < 2; ++kf) {
    int kb = kf * 32 + half * 8;
    const unsigned short* p = Qb + (size_t)(b * SEQ + s0 + l15) * D_MODEL + h * HEAD_DIM + kb;
    qf[kf].q[0] = *(const U4*)p;
    qf[kf].q[1] = *(const U4*)(p + 16);
  }

  f32x8 SA[16];
  for (int i = 0; i < 16; ++i) SA[i] = {};
#pragma unroll
  for (int nt = 0; nt < 16; ++nt) {
#pragma unroll
    for (int kf = 0; kf < 2; ++kf) {
      FragB bfr;
      int acol = nt * 16 + l15;
      int kb   = kf * 32 + half * 8;
      const unsigned short* p = agBase + (size_t)acol * HEAD_DIM + kb;
      bfr.q[0] = *(const U4*)p;
      bfr.q[1] = *(const U4*)(p + 16);
      SA[nt] = wmma_bf16(qf[kf], bfr, SA[nt]);
    }
  }

#pragma unroll
  for (int j = 0; j < 8; ++j) {
    float mx = -INFINITY;
#pragma unroll
    for (int nt = 0; nt < 16; ++nt) mx = fmaxf(mx, SA[nt][j] * scale);
    for (int msk = 1; msk < 16; msk <<= 1) mx = fmaxf(mx, __shfl_xor(mx, msk, 32));
    float p[16], sum = 0.f;
#pragma unroll
    for (int nt = 0; nt < 16; ++nt) { p[nt] = __expf(SA[nt][j] * scale - mx); sum += p[nt]; }
    for (int msk = 1; msk < 16; msk <<= 1) sum += __shfl_xor(sum, msk, 32);
    float inv  = 1.0f / sum;
    int   srow = s0 + j + half * 8;
    int   prow = j + half * 8;
#pragma unroll
    for (int nt = 0; nt < 16; ++nt) {
      float pn  = p[nt] * inv;
      int  acol = nt * 16 + l15;
      q_attn[((size_t)(b * N_HEAD + h) * SEQ + srow) * AGENTS + acol] = pn;
      Plds[prow][acol] = f2bf(pn);
    }
  }
  __syncthreads();

  f32x8 O[4];
  for (int i = 0; i < 4; ++i) O[i] = {};
#pragma unroll
  for (int kc = 0; kc < 8; ++kc) {
    FragB pf;
    int kb = kc * 32 + half * 8;
    pf.q[0] = *(const U4*)&Plds[l15][kb];
    pf.q[1] = *(const U4*)&Plds[l15][kb + 16];
#pragma unroll
    for (int nt = 0; nt < 4; ++nt) {
      FragB vb;
      int dimc  = nt * 16 + l15;
      int kbase = kc * 32 + half * 8;
#pragma unroll
      for (int e = 0; e < 16; ++e)
        vb.h[e] = avBase[(size_t)(kbase + (e & 7) + (e >> 3) * 16) * HEAD_DIM + dimc];
      O[nt] = wmma_bf16(pf, vb, O[nt]);
    }
  }

#pragma unroll
  for (int nt = 0; nt < 4; ++nt) {
    int dimc = nt * 16 + l15;
#pragma unroll
    for (int j = 0; j < 8; ++j) {
      int srow = s0 + j + half * 8;
      out_hb[(size_t)(b * SEQ + srow) * D_MODEL + h * HEAD_DIM + dimc] = f2bf(O[nt][j]);
    }
  }
}

// ---------------------------------------------------------------------------
extern "C" void kernel_launch(void* const* d_in, const int* in_sizes, int n_in,
                              void* d_out, int out_size, void* d_ws, size_t ws_size,
                              hipStream_t stream) {
  (void)in_sizes; (void)n_in; (void)out_size; (void)ws_size;
  const float* q  = (const float*)d_in[0];
  const float* k  = (const float*)d_in[1];
  const float* v  = (const float*)d_in[2];
  const float* Wq = (const float*)d_in[3];
  const float* bq = (const float*)d_in[4];
  const float* Wk = (const float*)d_in[5];
  const float* bk = (const float*)d_in[6];
  const float* Wv = (const float*)d_in[7];
  const float* bv = (const float*)d_in[8];
  const float* Wo = (const float*)d_in[9];
  const float* bo = (const float*)d_in[10];

  const size_t MD = (size_t)BATCH * SEQ * D_MODEL;               // 8,388,608
  const size_t DD = (size_t)D_MODEL * D_MODEL;                   // 1,048,576
  const size_t AD = (size_t)BATCH * N_HEAD * AGENTS * HEAD_DIM;  // 1,048,576

  unsigned short* p = (unsigned short*)d_ws;
  unsigned short* xq_b = p; p += MD;
  unsigned short* xk_b = p; p += MD;
  unsigned short* xv_b = p; p += MD;
  unsigned short* wq_b = p; p += DD;
  unsigned short* wk_b = p; p += DD;
  unsigned short* wv_b = p; p += DD;
  unsigned short* wo_b = p; p += DD;
  unsigned short* Qp_b = p; p += MD;
  unsigned short* Kp_b = p; p += MD;
  unsigned short* Vp_b = p; p += MD;
  unsigned short* ag_b = p; p += AD;
  unsigned short* av_b = p; p += AD;
  unsigned short* oh_b = p; p += MD;   // ~130 MB total

  float* out   = (float*)d_out;        // [B,S,D]
  float* qattn = out + MD;             // [B,H,S,A]

  const int M = BATCH * SEQ, N = D_MODEL, K = D_MODEL;

  // bulk fp32 -> bf16 conversions
  to_bf16<<<(int)(MD / 4 + 255) / 256, 256, 0, stream>>>(q,  xq_b, (int)(MD / 4));
  to_bf16<<<(int)(MD / 4 + 255) / 256, 256, 0, stream>>>(k,  xk_b, (int)(MD / 4));
  to_bf16<<<(int)(MD / 4 + 255) / 256, 256, 0, stream>>>(v,  xv_b, (int)(MD / 4));
  to_bf16<<<(int)(DD / 4 + 255) / 256, 256, 0, stream>>>(Wq, wq_b, (int)(DD / 4));
  to_bf16<<<(int)(DD / 4 + 255) / 256, 256, 0, stream>>>(Wk, wk_b, (int)(DD / 4));
  to_bf16<<<(int)(DD / 4 + 255) / 256, 256, 0, stream>>>(Wv, wv_b, (int)(DD / 4));
  to_bf16<<<(int)(DD / 4 + 255) / 256, 256, 0, stream>>>(Wo, wo_b, (int)(DD / 4));

  dim3 gblk(N / 128, M / 128);   // (8, 64)
  gemm_bias_wmma<<<gblk, 256, 0, stream>>>(xq_b, wq_b, bq, Qp_b, 0, M, N, K);
  gemm_bias_wmma<<<gblk, 256, 0, stream>>>(xk_b, wk_b, bk, Kp_b, 0, M, N, K);
  gemm_bias_wmma<<<gblk, 256, 0, stream>>>(xv_b, wv_b, bv, Vp_b, 0, M, N, K);

  agent_pool<<<(int)(AD / 256), 256, 0, stream>>>(Qp_b, ag_b);

  stage1_agent_kv<<<dim3(AGENTS / 16, N_HEAD, BATCH), 32, 0, stream>>>(Kp_b, Vp_b, ag_b, av_b);
  stage2_q_agent<<<dim3(SEQ / 16, N_HEAD, BATCH), 32, 0, stream>>>(Qp_b, ag_b, av_b, qattn, oh_b);

  gemm_bias_wmma<<<gblk, 256, 0, stream>>>(oh_b, wo_b, bo, out, 1, M, N, K);
}
